// NeuralCDE_5574867550237
// MI455X (gfx1250) — matrix operand order repaired and checked by reading, not compiled, not run on hardware
//
#include <hip/hip_runtime.h>
#include <cstddef>

// ---------------------------------------------------------------------------
// CDNA5 (gfx1250, wave32) Neural-CDE implementation.
//   B=256, L=2049, C=16, HID=64, WID=128, OUT=10, STEP=4, N_STEPS=512
// Strategy:
//   k_pack   : fp32 weights -> bf16 WMMA B-fragment layout (L2-resident, 0.3MB)
//   k_init   : z0 = tanh-MLP(coeffs[:,0,:16])                 (tiny, scalar)
//   k_scan   : persistent 512-step Euler scan, 16 WGs x 16 batch rows,
//              all GEMMs via v_wmma_f32_16x16x32_bf16, state in LDS
//   k_dec    : decoder over 256*2049 rows, one wave per 16-row tile, WMMA
// tanh is computed branch-free with v_exp_f32/v_rcp_f32 (no libm exec churn).
// ---------------------------------------------------------------------------

typedef __attribute__((ext_vector_type(16))) __bf16 v16bf;
typedef __attribute__((ext_vector_type(2)))  __bf16 v2bf;
typedef __attribute__((ext_vector_type(8)))  float  v8f;
typedef __attribute__((ext_vector_type(4)))  float  v4f;
typedef __attribute__((ext_vector_type(2)))  float  v2f;

union AFrag { v4f f[2]; v16bf v; };

// Branch-free tanh: tanh(x) = 1 - 2*rcp(exp(2x)+1), exp via v_exp_f32.
// Saturates to +/-1 through natural overflow/underflow; no EXEC divergence.
__device__ __forceinline__ float fast_tanh(float x) {
    float e = __builtin_amdgcn_exp2f(x * 2.885390081777927f);   // exp(2x)
    return 1.0f - 2.0f * __builtin_amdgcn_rcpf(e + 1.0f);
}

// A-matrix (16x32 bf16) fragment from LDS, row-major [16][ld] activation tile.
// ISA 7.12.2: lane m=l%16, half=l/16; elems 0..7 -> K=kt*32+8*half+{0..7},
// elems 8..15 -> +16.  Both runs are 16B-contiguous -> two ds_load_b128.
__device__ __forceinline__ v16bf load_afrag(const __bf16* base, int ld, int kt, int lane) {
    const int m = lane & 15, half = lane >> 4;
    const int k0 = kt * 32 + half * 8;
    AFrag u;
    u.f[0] = *(const v4f*)(base + m * ld + k0);
    u.f[1] = *(const v4f*)(base + m * ld + k0 + 16);
    return u.v;
}

// B-matrix fragment: pre-packed so each lane's 16 bf16 are contiguous (32B).
__device__ __forceinline__ v16bf load_bfrag(const __bf16* __restrict__ packed, int frag, int lane) {
    return *(const v16bf*)(packed + ((size_t)frag * 32 + lane) * 16);
}

__device__ __forceinline__ v8f wmma_bf16(v16bf a, v16bf b, v8f c) {
    return __builtin_amdgcn_wmma_f32_16x16x32_bf16(false, a, false, b, (short)0, c, false, false);
}

// reduce across each 16-lane half of the wave32 (masks < 16 stay in-half)
__device__ __forceinline__ float red16(float p) {
    p += __shfl_xor(p, 8, 32);
    p += __shfl_xor(p, 4, 32);
    p += __shfl_xor(p, 2, 32);
    p += __shfl_xor(p, 1, 32);
    return p;
}

// ---------------------------------------------------------------------------
// Weight packer: W (K x N, row-major fp32, y = x @ W) -> bf16 B-fragments.
// frag(kt,nt): lane l, elem e  <->  W[kt*32 + (l/16)*8 + (e%8) + (e/8)*16][nt*16 + l%16]
// (B 32x16 layout inferred from ISA 7.12.4 sparse-B pattern). Pads with 0.
// ---------------------------------------------------------------------------
__global__ void k_pack(const float* __restrict__ W, __bf16* __restrict__ dst,
                       int K, int N, int NT, int total) {
    int idx = blockIdx.x * 256 + threadIdx.x;
    if (idx >= total) return;
    int e  = idx & 15;
    int l  = (idx >> 4) & 31;
    int f  = idx >> 9;
    int nt = f % NT, kt = f / NT;
    int k  = kt * 32 + (l >> 4) * 8 + (e & 7) + (e >> 3) * 16;
    int n  = nt * 16 + (l & 15);
    float v = (k < K && n < N) ? W[(size_t)k * N + n] : 0.0f;
    dst[idx] = (__bf16)v;
}

// ---------------------------------------------------------------------------
// z0 = tanh-MLP([16->128->128->64]) on coeffs[:,0,:16].  One block per batch
// row, one thread per neuron. ~10 MFLOP total -> not worth WMMA.
// ---------------------------------------------------------------------------
__global__ void __launch_bounds__(128)
k_init(const float* __restrict__ coeffs,
       const float* __restrict__ W0, const float* __restrict__ b0,
       const float* __restrict__ W1, const float* __restrict__ b1,
       const float* __restrict__ W2, const float* __restrict__ b2,
       float* __restrict__ z_nodes) {
    const int b = blockIdx.x;        // 0..255
    const int t = threadIdx.x;       // 0..127
    __shared__ float x[16], h1[128], h2[128];
    if (t < 16) x[t] = coeffs[(size_t)b * 2048 * 64 + t];
    __syncthreads();
    float s = b0[t];
    #pragma unroll
    for (int i = 0; i < 16; ++i) s += x[i] * W0[i * 128 + t];
    h1[t] = fast_tanh(s);
    __syncthreads();
    s = b1[t];
    for (int i = 0; i < 128; ++i) s += h1[i] * W1[i * 128 + t];
    h2[t] = fast_tanh(s);
    __syncthreads();
    if (t < 64) {
        s = b2[t];
        for (int i = 0; i < 128; ++i) s += h2[i] * W2[i * 64 + t];
        z_nodes[(size_t)b * 64 + t] = fast_tanh(s);   // node 0
    }
}

// ---------------------------------------------------------------------------
// Persistent Euler scan. 16 blocks x 256 threads (8 waves). Each block owns a
// 16-row batch tile. Per step:
//   L1: (16x64)@(64x128)  -> 2 WMMA / wave   (wave = N-tile)
//   L2: (16x128)@(128x128)-> 4 WMMA / wave
//   L3: (16x128)@(128x1024)->32 WMMA / wave  (8 N-tiles, each = one hidden h)
//   einsum over c=16 channels = in-register mul + 16-lane shfl reduction,
//   z update in LDS + node store to global.
// frac==0 in the reference (t multiples of 4), so dX = coeffs[:,4s,16:32].
// ---------------------------------------------------------------------------
__global__ void __launch_bounds__(256)
k_scan(const float* __restrict__ coeffs,
       const __bf16* __restrict__ W0p, const float* __restrict__ b0,
       const __bf16* __restrict__ W1p, const float* __restrict__ b1,
       const __bf16* __restrict__ W2p, const float* __restrict__ b2,
       float* __restrict__ z_nodes) {
    const int lane  = threadIdx.x & 31;
    const int wave  = threadIdx.x >> 5;
    const int half  = lane >> 4;
    const int col16 = lane & 15;
    const int bbase = blockIdx.x * 16;

    __shared__ __align__(16) float  zc[16][64];    // fp32 state
    __shared__ __align__(16) __bf16 zb[16][64];    // bf16 copy for WMMA A
    __shared__ __align__(16) __bf16 a1[16][128];
    __shared__ __align__(16) __bf16 a2[16][128];
    __shared__ __align__(16) float  dx[16][16];

    for (int i = threadIdx.x; i < 16 * 64; i += 256)
        ((float*)zc)[i] = z_nodes[(size_t)(bbase + (i >> 6)) * 64 + (i & 63)];
    __syncthreads();

    for (int s = 0; s < 512; ++s) {
        // fp32 state -> bf16 A-tile, packed pairs (v_cvt_pk_bf16 + ds_store_b32)
        {
            int i = threadIdx.x;                    // 512 pairs / 256 threads
            #pragma unroll
            for (int rep = 0; rep < 2; ++rep, i += 256) {
                v2f f = ((const v2f*)zc)[i];
                v2bf p; p.x = (__bf16)f.x; p.y = (__bf16)f.y;
                ((v2bf*)zb)[i] = p;
            }
        }
        {   // dX tile: coeffs[bbase+m, 4s, 16+c]
            int m = threadIdx.x >> 4, c = threadIdx.x & 15;
            dx[m][c] = coeffs[((size_t)(bbase + m) * 2048 + (size_t)s * 4) * 64 + 16 + c];
        }
        __syncthreads();

        // ---- layer 1 ----
        {
            v8f acc; float bias = b0[wave * 16 + col16];
            #pragma unroll
            for (int r = 0; r < 8; ++r) acc[r] = bias;
            #pragma unroll
            for (int kt = 0; kt < 2; ++kt)
                acc = wmma_bf16(load_afrag(&zb[0][0], 64, kt, lane),
                                load_bfrag(W0p, kt * 8 + wave, lane), acc);
            int colb = wave * 16 + col16;
            #pragma unroll
            for (int r = 0; r < 8; ++r)
                a1[r + half * 8][colb] = (__bf16)fast_tanh(acc[r]);
        }
        __syncthreads();

        // ---- layer 2 ----
        {
            v8f acc; float bias = b1[wave * 16 + col16];
            #pragma unroll
            for (int r = 0; r < 8; ++r) acc[r] = bias;
            #pragma unroll
            for (int kt = 0; kt < 4; ++kt)
                acc = wmma_bf16(load_afrag(&a1[0][0], 128, kt, lane),
                                load_bfrag(W1p, kt * 8 + wave, lane), acc);
            int colb = wave * 16 + col16;
            #pragma unroll
            for (int r = 0; r < 8; ++r)
                a2[r + half * 8][colb] = (__bf16)fast_tanh(acc[r]);
        }
        __syncthreads();

        // ---- layer 3 + channel contraction + Euler update ----
        float dxv[8];
        #pragma unroll
        for (int r = 0; r < 8; ++r) dxv[r] = dx[half * 8 + r][col16];
        v16bf af[4];
        #pragma unroll
        for (int kt = 0; kt < 4; ++kt) af[kt] = load_afrag(&a2[0][0], 128, kt, lane);

        #pragma unroll
        for (int i = 0; i < 8; ++i) {
            const int h = wave * 8 + i;            // N-tile == hidden unit h
            v8f acc; float bias = b2[h * 16 + col16];
            #pragma unroll
            for (int r = 0; r < 8; ++r) acc[r] = bias;
            #pragma unroll
            for (int kt = 0; kt < 4; ++kt)
                acc = wmma_bf16(af[kt], load_bfrag(W2p, kt * 64 + h, lane), acc);
            #pragma unroll
            for (int r = 0; r < 8; ++r) {
                float ssum = red16(acc[r] * dxv[r]);   // sum over c = lanes
                if (col16 == 0) {                      // lanes 0 & 16 commit
                    int m = r + half * 8;
                    float znew = zc[m][h] + 4.0f * ssum;   // STEP = 4
                    zc[m][h] = znew;
                    z_nodes[((size_t)(s + 1) * 256 + bbase + m) * 64 + h] = znew;
                }
            }
        }
        __syncthreads();
    }
}

// ---------------------------------------------------------------------------
// Decoder: out[b,t,:] = tanh(z_all @ W0 + b0) @ W1 + b1, W1 N-padded 10->16.
// One wave per 16-row (b, t0..t0+15) tile; 8 tiles/block; 256*129 tiles exact.
// ---------------------------------------------------------------------------
__global__ void __launch_bounds__(256)
k_dec(const float* __restrict__ z_nodes,
      const __bf16* __restrict__ W0p, const float* __restrict__ b0,
      const __bf16* __restrict__ W1p, const float* __restrict__ b1,
      float* __restrict__ out) {
    const int lane  = threadIdx.x & 31;
    const int wave  = threadIdx.x >> 5;
    const int half  = lane >> 4;
    const int col16 = lane & 15;

    __shared__ __align__(16) __bf16 zb[8][16][64];
    __shared__ __align__(16) __bf16 h1[8][16][64];

    const int tile = blockIdx.x * 8 + wave;   // grid sized so this is exact
    const int b  = tile / 129;
    const int t0 = (tile % 129) * 16;

    {   // linear interpolation of z at times t0..t0+15 (row m handled by 2 lanes)
        int m = col16;
        int t = t0 + m; if (t > 2048) t = 2048;
        int j = t >> 2; if (j > 511) j = 511;
        float w = (float)(t - j * 4) * 0.25f;
        const float* z0p = &z_nodes[((size_t)j * 256 + b) * 64];
        const float* z1p = z0p + 256 * 64;
        #pragma unroll
        for (int cc = 0; cc < 32; cc += 2) {
            int cl = half * 32 + cc;
            v2f za = *(const v2f*)(z0p + cl);
            v2f zbv = *(const v2f*)(z1p + cl);
            v2bf p;
            p.x = (__bf16)(za.x + w * (zbv.x - za.x));
            p.y = (__bf16)(za.y + w * (zbv.y - za.y));
            *(v2bf*)&zb[wave][m][cl] = p;
        }
    }
    __syncthreads();

    v16bf a0  = load_afrag(&zb[wave][0][0], 64, 0, lane);
    v16bf a1f = load_afrag(&zb[wave][0][0], 64, 1, lane);
    #pragma unroll
    for (int nt = 0; nt < 4; ++nt) {
        v8f acc; float bias = b0[nt * 16 + col16];
        #pragma unroll
        for (int r = 0; r < 8; ++r) acc[r] = bias;
        acc = wmma_bf16(a0,  load_bfrag(W0p, 0 * 4 + nt, lane), acc);
        acc = wmma_bf16(a1f, load_bfrag(W0p, 1 * 4 + nt, lane), acc);
        #pragma unroll
        for (int r = 0; r < 8; ++r)
            h1[wave][r + half * 8][nt * 16 + col16] = (__bf16)fast_tanh(acc[r]);
    }
    __syncthreads();

    {
        v8f acc; float bias = (col16 < 10) ? b1[col16] : 0.0f;
        #pragma unroll
        for (int r = 0; r < 8; ++r) acc[r] = bias;
        #pragma unroll
        for (int kt = 0; kt < 2; ++kt)
            acc = wmma_bf16(load_afrag(&h1[wave][0][0], 64, kt, lane),
                            load_bfrag(W1p, kt, lane), acc);
        #pragma unroll
        for (int r = 0; r < 8; ++r) {
            int tt = t0 + r + half * 8;
            if (tt <= 2048 && col16 < 10)
                out[((size_t)b * 2049 + tt) * 10 + col16] = acc[r];
        }
    }
}

// ---------------------------------------------------------------------------
extern "C" void kernel_launch(void* const* d_in, const int* in_sizes, int n_in,
                              void* d_out, int out_size, void* d_ws, size_t ws_size,
                              hipStream_t stream) {
    const float* coeffs = (const float*)d_in[0];
    const float* iW0 = (const float*)d_in[1];  const float* ib0 = (const float*)d_in[2];
    const float* iW1 = (const float*)d_in[3];  const float* ib1 = (const float*)d_in[4];
    const float* iW2 = (const float*)d_in[5];  const float* ib2 = (const float*)d_in[6];
    const float* fW0 = (const float*)d_in[7];  const float* fb0 = (const float*)d_in[8];
    const float* fW1 = (const float*)d_in[9];  const float* fb1 = (const float*)d_in[10];
    const float* fW2 = (const float*)d_in[11]; const float* fb2 = (const float*)d_in[12];
    const float* dW0 = (const float*)d_in[13]; const float* db0 = (const float*)d_in[14];
    const float* dW1 = (const float*)d_in[15]; const float* db1 = (const float*)d_in[16];
    float* outp = (float*)d_out;

    // workspace carve-up: z_nodes (513*256*64 f32) then packed bf16 weights
    float*  z_nodes = (float*)d_ws;
    const size_t ZN = (size_t)513 * 256 * 64;
    __bf16* fW0p = (__bf16*)((char*)d_ws + ZN * sizeof(float));
    __bf16* fW1p = fW0p + 8192;     // 2kt*8nt*512
    __bf16* fW2p = fW1p + 16384;    // 4kt*8nt*512
    __bf16* dW0p = fW2p + 131072;   // 4kt*64nt*512
    __bf16* dW1p = dW0p + 4096;     // 2kt*4nt*512

    k_pack<<<32,  256, 0, stream>>>(fW0, fW0p,  64,  128,  8,   8192);
    k_pack<<<64,  256, 0, stream>>>(fW1, fW1p, 128,  128,  8,  16384);
    k_pack<<<512, 256, 0, stream>>>(fW2, fW2p, 128, 1024, 64, 131072);
    k_pack<<<16,  256, 0, stream>>>(dW0, dW0p,  64,   64,  4,   4096);
    k_pack<<<4,   256, 0, stream>>>(dW1, dW1p,  64,   10,  1,   1024);

    k_init<<<256, 128, 0, stream>>>(coeffs, iW0, ib0, iW1, ib1, iW2, ib2, z_nodes);
    k_scan<<<16,  256, 0, stream>>>(coeffs, fW0p, fb0, fW1p, fb1, fW2p, fb2, z_nodes);
    k_dec <<<4128,256, 0, stream>>>(z_nodes, dW0p, db0, dW1p, db1, outp);
}
